// cmdnet_bin_kb_36936718745663
// MI455X (gfx1250) — matrix operand order repaired and checked by reading, not compiled, not run on hardware
//
#include <hip/hip_runtime.h>
#include <hip/hip_bf16.h>

typedef __attribute__((ext_vector_type(16))) __bf16 v16bf;
typedef __attribute__((ext_vector_type(8)))  float  v8f;

#define NR_    128
#define NT_    64
#define NITER_ 64
#define HT_PITCH 136   // padded r-pitch for transposed bf16 Ht in LDS (avoids bank conflicts)
#define HH_PITCH 66    // padded pitch for f32 HH in LDS

__device__ __forceinline__ unsigned short f32_to_bf16_rn(float f) {
  unsigned u = __float_as_uint(f);
  unsigned r = u + 0x7FFFu + ((u >> 16) & 1u);   // round-to-nearest-even
  return (unsigned short)(r >> 16);
}
__device__ __forceinline__ float bf16_to_f32(unsigned short h) {
  return __uint_as_float(((unsigned)h) << 16);
}

__global__ __launch_bounds__(256)
void cmdnet_bin_kernel(const float* __restrict__ yt, const float* __restrict__ Ht,
                       const float* __restrict__ sigmat0, const float* __restrict__ alpha,
                       const float* __restrict__ taui, const float* __restrict__ delta,
                       float* __restrict__ out, int B) {
  __shared__ unsigned short sHtT[NT_ * HT_PITCH]; // Ht[b] transposed, bf16: [n][r]
  __shared__ float sHH[NT_ * HH_PITCH];           // Gram matrix f32: [n][m]
  __shared__ float sYt[NR_];
  __shared__ float sYH[NT_];
  __shared__ float sC[NT_];
  __shared__ float sS[NT_];
  __shared__ float sXt[NT_];
  __shared__ float sPart[4 * NT_];
  __shared__ float sTaui[NITER_];
  __shared__ float sDelta[NITER_];

  const int b    = blockIdx.x;
  const int tid  = threadIdx.x;
  const int lane = tid & 31;
  const int wave = tid >> 5;
  const int laneLo = lane & 15;
  const int hi     = (lane >> 4) & 1;

  const float* HtB = Ht + (size_t)b * NR_ * NT_;

  // ---- Stage: wide (b128) coalesced global reads of Ht[b], bf16-convert,
  //      transpose into LDS. 8192 f32 = 2048 float4 -> 8 loads per thread. ----
  {
    const float4* HtB4 = (const float4*)HtB;
    #pragma unroll
    for (int j = 0; j < (NR_ * NT_ / 4) / 256; ++j) {
      int i = tid + j * 256;
      float4 v = HtB4[i];
      int base = i << 2;
      int r = base >> 6;        // / NT_
      int n = base & 63;        // % NT_  (4 consecutive n, same r; 4 | 64 so no wrap)
      sHtT[(n + 0) * HT_PITCH + r] = f32_to_bf16_rn(v.x);
      sHtT[(n + 1) * HT_PITCH + r] = f32_to_bf16_rn(v.y);
      sHtT[(n + 2) * HT_PITCH + r] = f32_to_bf16_rn(v.z);
      sHtT[(n + 3) * HT_PITCH + r] = f32_to_bf16_rn(v.w);
    }
  }
  if (tid < NR_) sYt[tid] = yt[(size_t)b * NR_ + tid];
  if (tid < NT_) { sC[tid] = logf(1.0f / alpha[tid] - 1.0f); sS[tid] = 0.0f; }
  if (tid >= 64  && tid < 128) sTaui[tid - 64]   = taui[tid - 64];
  if (tid >= 128 && tid < 192) sDelta[tid - 128] = delta[tid - 128];
  __syncthreads();

  // ---- Gram: HH = Ht^T * Ht via V_WMMA_F32_16X16X32_BF16 (16 tiles, K=128).
  //      tn is wave-invariant across the 2 tiles (B fragments still reloaded per tile). ----
  union FragAB { v16bf v; unsigned u[8]; };
  for (int tile = wave; tile < 16; tile += 8) {   // 2 tiles per wave, full EXEC
    const int tm = tile >> 2, tn = tile & 3;
    const int nrow = tm * 16 + laneLo;   // A-matrix row (n index), M=lane%16
    const int mcol = tn * 16 + laneLo;   // B-matrix col (m index), N=lane%16
    v8f acc = {};
    #pragma unroll
    for (int kk = 0; kk < 4; ++kk) {
      const int r0 = kk * 32;
      FragAB a, bb;
      #pragma unroll
      for (int v = 0; v < 8; ++v) {
        // A 16x32 bf16 layout: VGPR v<4 -> K=2v(+8 if hi); v>=4 -> K=16+2(v-4)(+8 if hi)
        int aK = ((v < 4) ? (2 * v) : (16 + 2 * (v - 4))) + (hi ? 8 : 0);
        // B 32x16 bf16 layout: VGPR v<4 -> K=2v(+8 if hi); v>=4 -> K=16+2(v-4)(+8 if hi), N = lane%16
        int bK = ((v < 4) ? (2 * v) : (16 + 2 * (v - 4))) + (hi ? 8 : 0);
        a.u[v]  = *(const unsigned*)&sHtT[nrow * HT_PITCH + r0 + aK];
        bb.u[v] = *(const unsigned*)&sHtT[mcol * HT_PITCH + r0 + bK];
      }
      acc = __builtin_amdgcn_wmma_f32_16x16x32_bf16(false, a.v, false, bb.v,
                                                    (short)0, acc, false, false);
    }
    // f32 C/D layout: VGPR v -> row v (lanes 0-15) / row v+8 (lanes 16-31), col = lane%16
    #pragma unroll
    for (int v = 0; v < 8; ++v) {
      int row = tm * 16 + v + (hi ? 8 : 0);
      int col = tn * 16 + laneLo;
      sHH[row * HH_PITCH + col] = acc[v];
    }
  }

  // ---- yH[n] = sum_r yt[r]*Ht[r][n] (from resident bf16 copy, f32 accumulate;
  //      lowers to packed v_fma_mix_f32_bf16 over ds_load_b128) ----
  if (tid < NT_) {
    float accY = 0.0f;
    for (int r = 0; r < NR_; ++r)
      accY = __builtin_fmaf(sYt[r], bf16_to_f32(sHtT[tid * HT_PITCH + r]), accY);
    sYH[tid] = accY;
  }
  __syncthreads();

  const float sg   = sigmat0[b];
  const float sig2 = sg * sg;
  const int m    = tid & 63;
  const int part = tid >> 6;   // 4-way split of the n-sum

  // ---- 64 sequential CMD iterations, fully LDS-resident (zero HBM traffic) ----
  for (int it = 0; it < NITER_; ++it) {
    float ta = fabsf(sTaui[it]);
    float tx = (it == 0) ? 1.0f : ta;
    if (tid < NT_) sXt[tid] = tanhf((sC[tid] + sS[tid]) * 0.5f * tx);
    __syncthreads();
    float p = 0.0f;
    #pragma unroll
    for (int nn = 0; nn < 16; ++nn) {
      int n = part * 16 + nn;
      p = __builtin_fmaf(sXt[n], sHH[n * HH_PITCH + m], p);
    }
    sPart[part * NT_ + m] = p;
    __syncthreads();
    if (tid < NT_) {
      float xhh = sPart[tid] + sPart[NT_ + tid] + sPart[2 * NT_ + tid] + sPart[3 * NT_ + tid];
      float xv = sXt[tid];
      float sv = sS[tid];
      float grad = sig2 * tanhf(sv * 0.5f)
                 + 0.5f * ta * (1.0f - xv * xv) * (xhh - sYH[tid]);
      sS[tid] = sv - sDelta[it] * grad;
    }
    __syncthreads();
  }

  // ---- Final soft outputs: ft [B,NT,2] then xt [B,NT], concatenated flat ----
  if (tid < NT_) {
    float ta = fabsf(sTaui[NITER_ - 1]);
    float xv = tanhf((sC[tid] + sS[tid]) * 0.5f * ta);
    size_t base = (size_t)b * NT_ + tid;
    out[base * 2 + 0] = (1.0f - xv) * 0.5f;
    out[base * 2 + 1] = (1.0f + xv) * 0.5f;
    out[(size_t)B * NT_ * 2 + base] = xv;
  }
}

extern "C" void kernel_launch(void* const* d_in, const int* in_sizes, int n_in,
                              void* d_out, int out_size, void* d_ws, size_t ws_size,
                              hipStream_t stream) {
  const float* yt      = (const float*)d_in[0];
  const float* Ht      = (const float*)d_in[1];
  const float* sigmat0 = (const float*)d_in[2];
  const float* alpha   = (const float*)d_in[3];
  const float* taui    = (const float*)d_in[4];
  const float* delta   = (const float*)d_in[5];
  const int B = in_sizes[2];   // sigmat0 has B elements

  cmdnet_bin_kernel<<<B, 256, 0, stream>>>(yt, Ht, sigmat0, alpha, taui, delta,
                                           (float*)d_out, B);
}
